// NTXentLoss_49357764166268
// MI455X (gfx1250) — matrix operand order repaired and checked by reading, not compiled, not run on hardware
//
#include <hip/hip_runtime.h>
#include <hip/hip_bf16.h>

// ---------------------------------------------------------------------------
// NT-Xent loss, fused for MI455X (gfx1250, wave32, WMMA).
//   B=4096, D=128, TEMP=0.5 -> logits = 2*sim, bounded by ~2 (cosine sims).
//   loss = mean_i( lse_i - pos_i ),  lse over row of (2*sim) with self masked.
//   Fixed-shift logsumexp: lse_i = 2 + log( sum_j exp(2*sim_ij - 2) ) with the
//   self term zeroed (exact equivalent of the reference -inf mask).
// Pipeline:
//   K1: row-normalize z=concat(zi,zj) in fp32, store zn as f16 (2 MB, L2-hot)
//   K2: 512 WGs x 4 waves; each WG owns 16 rows, waves stride 512 col tiles,
//       v_wmma_f32_16x16x32_f16 (4 per tile, K=128) with double-buffered B
//       operands so loads for tile i+1 overlap the WMMAs/exp of tile i.
//   K3: reduce 8192 per-row partials -> d_out[0]
// ---------------------------------------------------------------------------

typedef _Float16 v16h __attribute__((ext_vector_type(16)));
typedef _Float16 v8h  __attribute__((ext_vector_type(8)));
typedef float    v8f  __attribute__((ext_vector_type(8)));

#define NB    4096          // batch B
#define N2    8192          // 2B rows
#define DD    128           // feature dim
#define NTILE (N2 / 16)     // 512 column tiles

// ---- K1: normalize rows of concat(zi,zj), write f16 zn -------------------
__global__ __launch_bounds__(256) void ntx_normalize(const float* __restrict__ zi,
                                                     const float* __restrict__ zj,
                                                     _Float16* __restrict__ znh) {
    const int wave = threadIdx.x >> 5;
    const int lane = threadIdx.x & 31;
    const int row  = blockIdx.x * 8 + wave;           // grid = N2/8 blocks
    const float* src = (row < NB) ? (zi + (size_t)row * DD)
                                  : (zj + (size_t)(row - NB) * DD);
    float4 v = *(const float4*)(src + lane * 4);      // 4 floats per lane
    float ss = v.x * v.x + v.y * v.y + v.z * v.z + v.w * v.w;
#pragma unroll
    for (int off = 16; off >= 1; off >>= 1) ss += __shfl_xor(ss, off, 32);
    const float inv = 1.0f / fmaxf(sqrtf(ss), 1e-8f); // torch clamp at EPS
    _Float16* dst = znh + (size_t)row * DD + lane * 4;
    dst[0] = (_Float16)(v.x * inv);
    dst[1] = (_Float16)(v.y * inv);
    dst[2] = (_Float16)(v.z * inv);
    dst[3] = (_Float16)(v.w * inv);
}

__device__ __forceinline__ const _Float16* b_addr(const _Float16* znh, int ct,
                                                  int l16, bool hi) {
    // B operand 32x16 layout: lanes 0-15 hold K=k0..k0+15 of column l16,
    // lanes 16-31 hold K=k0+16..k0+31 (contiguous 32B per lane).
    return znh + (size_t)(ct * 16 + l16) * DD + (hi ? 16 : 0);
}

// ---- K2: fused sim + masked exp-sum + positive extraction ----------------
__global__ __launch_bounds__(128) void ntx_main(const _Float16* __restrict__ znh,
                                                float* __restrict__ partial) {
    const int tid   = threadIdx.x;
    const int wave  = tid >> 5;
    const int lane  = tid & 31;
    const int l16   = lane & 15;
    const bool hi   = (lane >= 16);
    const int rbase = blockIdx.x * 16;

    // A operand: rows rbase+l16, all of K=128, resident in 32 VGPRs.
    // A 16x32 layout (contiguous, matching B): lanes 0-15 hold K = k0+{0..15},
    // lanes 16-31 hold K = k0+{16..31}.
    v16h a[4];
    {
        const _Float16* arow = znh + (size_t)(rbase + l16) * DD;
#pragma unroll
        for (int kc = 0; kc < 4; ++kc) {
            const int k0 = kc * 32 + (hi ? 16 : 0);
            v16h t;
            *((v8h*)&t)     = *(const v8h*)(arow + k0);       // K=k0..k0+7
            *((v8h*)&t + 1) = *(const v8h*)(arow + k0 + 8);   // K=k0+8..+15
            a[kc] = t;
        }
    }

    float acc[8], pacc[8];
#pragma unroll
    for (int r = 0; r < 8; ++r) { acc[r] = 0.0f; pacc[r] = 0.0f; }

    const int myrow0 = rbase + (hi ? 8 : 0);  // C layout: row = myrow0 + r

    // ---- software-pipelined column-tile loop (uniform per wave) ----------
    int ct = wave;
    const _Float16* bp = b_addr(znh, ct, l16, hi);
    v16h b0 = *(const v16h*)(bp +  0);
    v16h b1 = *(const v16h*)(bp + 32);
    v16h b2 = *(const v16h*)(bp + 64);
    v16h b3 = *(const v16h*)(bp + 96);

    for (int it = 0; it < NTILE / 4; ++it) {
        // Issue next tile's loads first so they overlap WMMA + exp below.
        // Wrap the last prefetch to a valid tile (wave-uniform, no EXEC games).
        const int ctn = (it + 1 < NTILE / 4) ? ct + 4 : wave;
        const _Float16* bn = b_addr(znh, ctn, l16, hi);
        v16h n0 = *(const v16h*)(bn +  0);
        v16h n1 = *(const v16h*)(bn + 32);
        v16h n2 = *(const v16h*)(bn + 64);
        v16h n3 = *(const v16h*)(bn + 96);

        v8f c = {};
        c = __builtin_amdgcn_wmma_f32_16x16x32_f16(false, a[0], false, b0,
                                                   (short)0, c, false, false);
        c = __builtin_amdgcn_wmma_f32_16x16x32_f16(false, a[1], false, b1,
                                                   (short)0, c, false, false);
        c = __builtin_amdgcn_wmma_f32_16x16x32_f16(false, a[2], false, b2,
                                                   (short)0, c, false, false);
        c = __builtin_amdgcn_wmma_f32_16x16x32_f16(false, a[3], false, b3,
                                                   (short)0, c, false, false);

        const int col = ct * 16 + l16;  // C layout: N = lane % 16
#pragma unroll
        for (int r = 0; r < 8; ++r) {
            const int   row     = myrow0 + r;
            const int   partner = row + ((row < NB) ? NB : -NB);
            const float logit   = 2.0f * c[r];            // sim / TEMP
            const float e       = __expf(logit - 2.0f);   // fixed-shift LSE
            acc[r]  += (col == row)     ? 0.0f  : e;      // mask self-diag
            pacc[r] += (col == partner) ? logit : 0.0f;   // positive logit
        }

        b0 = n0; b1 = n1; b2 = n2; b3 = n3;
        ct += 4;
    }

    // Reduce across the 16 lanes of each half-wave (xor 8..1 stays in-half).
#pragma unroll
    for (int r = 0; r < 8; ++r) {
#pragma unroll
        for (int off = 8; off >= 1; off >>= 1) {
            acc[r]  += __shfl_xor(acc[r],  off, 32);
            pacc[r] += __shfl_xor(pacc[r], off, 32);
        }
    }

    __shared__ float s_acc[4][16];
    __shared__ float s_pos[4][16];
    if (l16 == 0) {                 // lanes 0 and 16 of each wave
        const int base = hi ? 8 : 0;
#pragma unroll
        for (int r = 0; r < 8; ++r) {
            s_acc[wave][base + r] = acc[r];
            s_pos[wave][base + r] = pacc[r];
        }
    }
    __syncthreads();
    if (tid < 16) {
        const float S = s_acc[0][tid] + s_acc[1][tid] + s_acc[2][tid] + s_acc[3][tid];
        const float P = s_pos[0][tid] + s_pos[1][tid] + s_pos[2][tid] + s_pos[3][tid];
        partial[rbase + tid] = 2.0f + __logf(S) - P;   // lse_row - pos_row
    }
}

// ---- K3: final scalar reduction ------------------------------------------
__global__ __launch_bounds__(256) void ntx_reduce(const float* __restrict__ partial,
                                                  float* __restrict__ out) {
    __shared__ float s[256];
    float sum = 0.0f;
    for (int i = threadIdx.x; i < N2; i += 256) sum += partial[i];
    s[threadIdx.x] = sum;
    __syncthreads();
#pragma unroll
    for (int off = 128; off >= 1; off >>= 1) {
        if (threadIdx.x < off) s[threadIdx.x] += s[threadIdx.x + off];
        __syncthreads();
    }
    if (threadIdx.x == 0) out[0] = s[0] / (float)N2;
}

extern "C" void kernel_launch(void* const* d_in, const int* in_sizes, int n_in,
                              void* d_out, int out_size, void* d_ws, size_t ws_size,
                              hipStream_t stream) {
    const float* zi = (const float*)d_in[0];
    const float* zj = (const float*)d_in[1];
    _Float16* znh   = (_Float16*)d_ws;                              // 2 MB
    float* partial  = (float*)((char*)d_ws + (size_t)N2 * DD * sizeof(_Float16));

    ntx_normalize<<<N2 / 8, 256, 0, stream>>>(zi, zj, znh);
    ntx_main<<<N2 / 16, 128, 0, stream>>>(znh, partial);
    ntx_reduce<<<1, 256, 0, stream>>>(partial, (float*)d_out);
}
